// RowLSTM_70789650972910
// MI455X (gfx1250) — matrix operand order repaired and checked by reading, not compiled, not run on hardware
//
#include <hip/hip_runtime.h>

// ---------------- problem constants ----------------
#define B_       32
#define CIN      3
#define H_       64
#define W_       64
#define HC       128
#define G4       512          // 4*HC gate channels
#define KTOT     256          // 2 taps * HC
#define NTHREADS 512          // 16 wave32
#define NWAVES   16

// LDS strides
#define HSTR  264             // HcatT row stride in halves (528 B = multiple of 16 -> ds_load_b128)
#define CSTR  65              // c buffer stride in floats  ([HC][CSTR])
#define GSTR  65              // gates buffer stride        ([G4][GSTR])

#define LDS_HCAT_B   (W_ * HSTR * 2)         // 33792
#define LDS_C_B      (HC * CSTR * 4)         // 33280
#define LDS_G_B      (G4 * GSTR * 4)         // 133120
#define LDS_WIS_B    (G4 * 7 * 4)            // 14336
#define LDS_XROW_B   (CIN * W_ * 4)          // 768
#define LDS_TOTAL_B  (LDS_HCAT_B + LDS_C_B + LDS_G_B + LDS_WIS_B + LDS_XROW_B)

typedef __attribute__((ext_vector_type(16))) _Float16 v16h;
typedef __attribute__((ext_vector_type(8)))  float    v8f;
typedef __attribute__((ext_vector_type(4)))  unsigned v4u;

__global__ __launch_bounds__(NTHREADS)
void rowlstm_kernel(const float* __restrict__ x,
                    const float* __restrict__ W_is,
                    const float* __restrict__ b_is,
                    const float* __restrict__ W_ss,
                    const float* __restrict__ b_ss,
                    const float* __restrict__ h0,
                    const float* __restrict__ c0,
                    float* __restrict__ out)
{
    extern __shared__ char smem[];
    _Float16* hcat = (_Float16*)smem;                         // [W_][HSTR]  im2col of h_prev (f16)
    float*    cbuf = (float*)(smem + LDS_HCAT_B);             // [HC][CSTR]  cell state (f32)
    float*    gbuf = (float*)(smem + LDS_HCAT_B + LDS_C_B);   // [G4][GSTR]  gate values (f32)
    float*    wis  = (float*)(smem + LDS_HCAT_B + LDS_C_B + LDS_G_B);           // [G4][7]
    float*    xrow = (float*)(smem + LDS_HCAT_B + LDS_C_B + LDS_G_B + LDS_WIS_B); // [CIN][W_]

    const int tid  = threadIdx.x;
    const int lane = tid & 31;
    const int wv   = tid >> 5;        // wave id 0..15, owns gate channels [32*wv, 32*wv+32)
    const int lo   = lane & 15;
    const int hi   = lane >> 4;
    const int b    = blockIdx.x;      // one workgroup per batch element

    // ---------------------------------------------------------------
    // Preload masked recurrent weights as WMMA A fragments (VGPR-resident
    // across all 64 row steps).  Effective A[oc][k]:
    //   k <  128 : left-tap   W_ss[oc][k][0][0]
    //   k >= 128 : center-tap W_ss[oc][k-128][0][1]   (tap 2 masked to 0)
    // f16 A fragment, k order chosen here (same as B): lane holds row M = lane&15;
//   VGPR vi pair-of-halves = K = 32*kt + (lane>=16?16:0) + 2*vi .. +1
    // ---------------------------------------------------------------
    v16h a[2][8];
    #pragma unroll
    for (int mi = 0; mi < 2; ++mi) {
        const int oc = (2 * wv + mi) * 16 + lo;
        const float* wrow = W_ss + (size_t)oc * (HC * 3);
        #pragma unroll
        for (int kt = 0; kt < 8; ++kt) {
            #pragma unroll
            for (int vi = 0; vi < 8; ++vi) {
                const int k0 = kt * 32 + hi * 16 + 2 * vi;
                #pragma unroll
                for (int s = 0; s < 2; ++s) {
                    const int k = k0 + s;
                    const float w = (k < HC) ? wrow[k * 3 + 0] : wrow[(k - HC) * 3 + 1];
                    a[mi][kt][2 * vi + s] = (_Float16)w;
                }
            }
        }
    }

    // ---- masked input-conv weights + fused (b_is + b_ss) bias into LDS ----
    for (int oc = tid; oc < G4; oc += NTHREADS) {
        #pragma unroll
        for (int ic = 0; ic < CIN; ++ic) {
            wis[oc * 7 + ic]     = W_is[oc * 9 + ic * 3 + 0];   // left tap
            wis[oc * 7 + 3 + ic] = W_is[oc * 9 + ic * 3 + 1];   // center tap
        }
        wis[oc * 7 + 6] = b_is[oc] + b_ss[oc];
    }
    // zero HcatT (left-pad column at n==0 must stay 0 forever)
    for (int i = tid; i < W_ * HSTR; i += NTHREADS) hcat[i] = (_Float16)0.f;
    __syncthreads();

    // broadcast h0 / c0 into LDS state
    for (int e = tid; e < HC * W_; e += NTHREADS) {
        const int ch = e >> 6, w = e & 63;
        const float hv = h0[ch * W_ + w];
        hcat[w * HSTR + HC + ch] = (_Float16)hv;                    // center tap column
        if (w + 1 < W_) hcat[(w + 1) * HSTR + ch] = (_Float16)hv;   // left tap column
        cbuf[ch * CSTR + w] = c0[ch * W_ + w];
    }
    __syncthreads();

    const float* xb   = x   + (size_t)b * (CIN * H_ * W_);
    float*       outb = out + (size_t)b * (HC * H_ * W_);

    for (int row = 0; row < H_; ++row) {
        // ---- stage this row's x slice; prefetch next row's ----
        if (tid < CIN * W_) {
            const int ic = tid >> 6, w = tid & 63;
            xrow[ic * W_ + w] = xb[(size_t)ic * (H_ * W_) + row * W_ + w];
            if (row + 1 < H_)
                __builtin_prefetch(&xb[(size_t)ic * (H_ * W_) + (row + 1) * W_ + w], 0, 3);
        }
        __syncthreads();

        // ---- gates pre-activation GEMM: [512 x 256] x [256 x 64] ----
        v8f acc[2][4] = {};
        #pragma unroll
        for (int nt = 0; nt < 4; ++nt) {
            const int n = nt * 16 + lo;
            #pragma unroll
            for (int kt = 0; kt < 8; ++kt) {
                // f16 B fragment, k order chosen here (same as A): lane holds col N = lane&15;
                // VGPR vi pair = K = 32*kt + (lane>=16?16:0) + 2*vi .. +1
                // 32 contiguous, 16B-aligned bytes -> two ds_load_b128
                const int kbase = kt * 32 + hi * 16;
                union { v16h h; v4u q[2]; } bb;
                const v4u* hp = (const v4u*)&hcat[n * HSTR + kbase];
                bb.q[0] = hp[0];
                bb.q[1] = hp[1];
                acc[0][nt] = __builtin_amdgcn_wmma_f32_16x16x32_f16(
                    false, a[0][kt], false, bb.h, (short)0, acc[0][nt], false, false);
                acc[1][nt] = __builtin_amdgcn_wmma_f32_16x16x32_f16(
                    false, a[1][kt], false, bb.h, (short)0, acc[1][nt], false, false);
            }
        }

        // ---- epilogue: fuse masked 1x3 input conv + bias, sigmoid, spill gates ----
        float xl[4][CIN], xc[4][CIN];
        #pragma unroll
        for (int nt = 0; nt < 4; ++nt) {
            const int n = nt * 16 + lo;
            #pragma unroll
            for (int ic = 0; ic < CIN; ++ic) {
                xc[nt][ic] = xrow[ic * W_ + n];
                xl[nt][ic] = (n > 0) ? xrow[ic * W_ + n - 1] : 0.f;
            }
        }
        #pragma unroll
        for (int mi = 0; mi < 2; ++mi) {
            #pragma unroll
            for (int r = 0; r < 8; ++r) {
                const int oc = (2 * wv + mi) * 16 + hi * 8 + r;   // C tile: M = r + 8*hi
                const float wl0 = wis[oc * 7 + 0], wl1 = wis[oc * 7 + 1], wl2 = wis[oc * 7 + 2];
                const float wc0 = wis[oc * 7 + 3], wc1 = wis[oc * 7 + 4], wc2 = wis[oc * 7 + 5];
                const float bias = wis[oc * 7 + 6];
                #pragma unroll
                for (int nt = 0; nt < 4; ++nt) {
                    float pre = acc[mi][nt][r] + bias
                              + wl0 * xl[nt][0] + wl1 * xl[nt][1] + wl2 * xl[nt][2]
                              + wc0 * xc[nt][0] + wc1 * xc[nt][1] + wc2 * xc[nt][2];
                    const float sg = 1.f / (1.f + __expf(-pre));
                    gbuf[oc * GSTR + nt * 16 + lo] = sg;
                }
            }
        }
        __syncthreads();

        // ---- LSTM state update (gates cross wave boundaries via LDS) ----
        #pragma unroll
        for (int j = 0; j < (HC * W_) / NTHREADS; ++j) {   // 16 elems/thread, coalesced in w
            const int e  = j * NTHREADS + tid;
            const int ch = e >> 6, w = e & 63;
            const float og = gbuf[(0 * HC + ch) * GSTR + w];
            const float fg = gbuf[(1 * HC + ch) * GSTR + w];
            const float ig = gbuf[(2 * HC + ch) * GSTR + w];
            const float gg = gbuf[(3 * HC + ch) * GSTR + w];
            const float cv = fg * cbuf[ch * CSTR + w] + ig * gg;
            cbuf[ch * CSTR + w] = cv;
            // saturating tanh (avoid inf/inf NaN)
            const float t  = __expf(-2.f * fabsf(cv));
            const float th = __builtin_copysignf((1.f - t) / (1.f + t), cv);
            const float hv = og * th;
            outb[(size_t)ch * (H_ * W_) + row * W_ + w] = hv;
            hcat[w * HSTR + HC + ch] = (_Float16)hv;                   // center tap
            if (w + 1 < W_) hcat[(w + 1) * HSTR + ch] = (_Float16)hv;  // left tap
        }
        __syncthreads();
    }
}

extern "C" void kernel_launch(void* const* d_in, const int* in_sizes, int n_in,
                              void* d_out, int out_size, void* d_ws, size_t ws_size,
                              hipStream_t stream) {
    (void)in_sizes; (void)n_in; (void)d_ws; (void)ws_size; (void)out_size;
    const float* x    = (const float*)d_in[0];
    const float* W_is = (const float*)d_in[1];
    const float* b_is = (const float*)d_in[2];
    const float* W_ss = (const float*)d_in[3];
    const float* b_ss = (const float*)d_in[4];
    const float* h0   = (const float*)d_in[5];
    const float* c0   = (const float*)d_in[6];
    float* out = (float*)d_out;

    // CDNA5 allows up to 320 KB LDS per workgroup; opt in for >64 KB dynamic LDS.
    (void)hipFuncSetAttribute((const void*)rowlstm_kernel,
                              hipFuncAttributeMaxDynamicSharedMemorySize,
                              LDS_TOTAL_B);

    rowlstm_kernel<<<dim3(B_), dim3(NTHREADS), LDS_TOTAL_B, stream>>>(
        x, W_is, b_is, W_ss, b_ss, h0, c0, out);
}
